// KarmaDock_57973468561689
// MI455X (gfx1250) — compile-verified, matrix operand order checked by reading, not run on hardware
//
#include <hip/hip_runtime.h>
#include <stdint.h>

#define NPv  2048
#define NLv  512
#define NN   2560
#define DD   128
#define BBv  8
#define EPPv 40960
#define ELLv 5120
#define EPLv 10240
#define EE   66560
#define NLAY 8
#define NREC 3
#define WPB  4
#define K1PAD 392   // 385 padded to multiple of 8 (16B-aligned bf16 chunks)

typedef __attribute__((ext_vector_type(16))) __bf16 v16bf;
typedef __attribute__((ext_vector_type(8)))  float  v8f;

union Frag { v16bf v; uint4 q[2]; };

__device__ __forceinline__ unsigned short f2bf(float x) {
    unsigned int u = __float_as_uint(x);
    unsigned int r = u + 0x7FFFu + ((u >> 16) & 1u);   // round-to-nearest-even
    return (unsigned short)(r >> 16);
}
__device__ __forceinline__ float sigmoidf_(float x) { return 1.f / (1.f + __expf(-x)); }
__device__ __forceinline__ float siluf_(float x)    { return x * sigmoidf_(x); }

__device__ __forceinline__ Frag ldfrag(const unsigned short* p0, const unsigned short* p1) {
    Frag f;
    f.q[0] = *(const uint4*)p0;
    f.q[1] = *(const uint4*)p1;
    return f;
}
__device__ __forceinline__ v8f wmma_bf16(const Frag& a, const Frag& b, v8f c) {
    return __builtin_amdgcn_wmma_f32_16x16x32_bf16(false, a.v, false, b.v, (short)0, c, false, false);
}

// ---------------------------------------------------------------- utility kernels

__global__ void zero_kernel(float* p, long n) {
    long i = (long)blockIdx.x * blockDim.x + threadIdx.x;
    if (i < n) p[i] = 0.f;
}

// dst[l][n][k(pad)] = (k<K) ? src[l][k][n] : 0   (f32 -> bf16 bits)
__global__ void transpose_bf16_kernel(const float* __restrict__ src, unsigned short* __restrict__ dst,
                                      int L, int K, int Nn, int Kpad) {
    long idx = (long)blockIdx.x * blockDim.x + threadIdx.x;
    long total = (long)L * Nn * Kpad;
    if (idx >= total) return;
    int k = (int)(idx % Kpad);
    long t = idx / Kpad;
    int n = (int)(t % Nn);
    int l = (int)(t / Nn);
    float v = (k < K) ? src[((long)l * K + k) * Nn + n] : 0.f;
    dst[idx] = f2bf(v);
}

__global__ void graphnorm_kernel(const float* __restrict__ pro, const float* __restrict__ lig,
                                 const float* __restrict__ gamma, const float* __restrict__ beta,
                                 float* __restrict__ h, float* __restrict__ h0,
                                 unsigned short* __restrict__ hb, unsigned short* __restrict__ h0b) {
    int b = blockIdx.x, col = threadIdx.x;           // 8 blocks x 128 threads
    const int PPB = NPv / BBv, LPB = NLv / BBv;      // 256 pro + 64 lig per graph
    float s = 0.f;
    for (int i = 0; i < PPB; i++) s += pro[(size_t)(b * PPB + i) * DD + col];
    for (int i = 0; i < LPB; i++) s += lig[(size_t)(b * LPB + i) * DD + col];
    float mean = s / (float)(PPB + LPB);
    float v = 0.f;
    for (int i = 0; i < PPB; i++) { float d = pro[(size_t)(b * PPB + i) * DD + col] - mean; v += d * d; }
    for (int i = 0; i < LPB; i++) { float d = lig[(size_t)(b * LPB + i) * DD + col] - mean; v += d * d; }
    float inv = rsqrtf(v / (float)(PPB + LPB) + 1e-5f);
    float g = gamma[col], be = beta[col];
    for (int i = 0; i < PPB; i++) {
        int n = b * PPB + i;
        float x = (pro[(size_t)n * DD + col] - mean) * inv * g + be;
        size_t o = (size_t)n * DD + col;
        h[o] = x; h0[o] = x;
        unsigned short xb = f2bf(x);
        hb[o] = xb; h0b[o] = xb;
    }
    for (int i = 0; i < LPB; i++) {
        int n = NPv + b * LPB + i;
        float x = (lig[(size_t)(b * LPB + i) * DD + col] - mean) * inv * g + be;
        size_t o = (size_t)n * DD + col;
        h[o] = x; h0[o] = x;
        unsigned short xb = f2bf(x);
        hb[o] = xb; h0b[o] = xb;
    }
}

__global__ void pos_init_kernel(const float* __restrict__ pro_xyz, const float* __restrict__ lig_pos,
                                float* __restrict__ pos) {
    int n = blockIdx.x * blockDim.x + threadIdx.x;
    if (n >= NN) return;
    for (int c = 0; c < 3; c++)
        pos[n * 3 + c] = (n < NPv) ? pro_xyz[n * 3 + c] : lig_pos[(n - NPv) * 3 + c];
}

__global__ void build_edges_kernel(const int* __restrict__ p2p, const int* __restrict__ l2l,
                                   const int* __restrict__ p2l,
                                   int* __restrict__ u, int* __restrict__ v, float* __restrict__ countf) {
    int i = blockIdx.x * blockDim.x + threadIdx.x;
    if (i >= EE) return;
    int uu, vv;
    if (i < EPPv)                       { uu = p2p[i];                 vv = p2p[EPPv + i]; }
    else if (i < EPPv + ELLv)           { int j = i - EPPv;            uu = l2l[j] + NPv;       vv = l2l[ELLv + j] + NPv; }
    else if (i < EPPv + ELLv + EPLv)    { int j = i - EPPv - ELLv;     uu = p2l[j];             vv = p2l[EPLv + j] + NPv; }
    else                                { int j = i - EPPv - ELLv - EPLv; uu = p2l[EPLv + j] + NPv; vv = p2l[j]; }
    u[i] = uu; v[i] = vv;
    atomicAdd(&countf[vv], 1.0f);
}

__global__ void edge_init_kernel(const float* __restrict__ p2ps, const float* __restrict__ l2ls,
                                 const float* __restrict__ W, const float* __restrict__ b,
                                 float* __restrict__ einit, unsigned short* __restrict__ eib,
                                 float* __restrict__ e, unsigned short* __restrict__ eb) {
    long idx = (long)blockIdx.x * blockDim.x + threadIdx.x;
    if (idx >= (long)EE * DD) return;
    int ei = (int)(idx / DD), col = (int)(idx % DD);
    float val = b[col];
    if (ei < EPPv) {
        const float* r = p2ps + (size_t)ei * 6;
        for (int k = 0; k < 6; k++) val += r[k] * W[k * DD + col];
    } else if (ei < EPPv + ELLv) {
        const float* r = l2ls + (size_t)(ei - EPPv) * 6;
        for (int k = 0; k < 6; k++) val += r[k] * W[k * DD + col];
    } else {
        val -= W[5 * DD + col];   // zeros with last feature = -1
    }
    einit[idx] = val;
    e[idx] = val;
    unsigned short vb = f2bf(val);
    eib[idx] = vb; eb[idx] = vb;
}

__global__ void cvt_agg_kernel(float* __restrict__ agg, unsigned short* __restrict__ aggb) {
    long idx = (long)blockIdx.x * blockDim.x + threadIdx.x;
    if (idx >= (long)NN * DD) return;
    aggb[idx] = f2bf(agg[idx]);
    agg[idx] = 0.f;     // ready for next layer
}

__global__ void pos_rmsd_kernel(float* __restrict__ pos, float* __restrict__ dsum,
                                const float* __restrict__ countf, const float* __restrict__ lig_xyz,
                                const int* __restrict__ lig_batch, float* __restrict__ racc) {
    int n = blockIdx.x * blockDim.x + threadIdx.x;
    if (n >= NN) return;
    if (n >= NPv) {
        float c = fmaxf(countf[n], 1.f);
        float px = pos[n * 3 + 0] + dsum[n * 3 + 0] / c;
        float py = pos[n * 3 + 1] + dsum[n * 3 + 1] / c;
        float pz = pos[n * 3 + 2] + dsum[n * 3 + 2] / c;
        pos[n * 3 + 0] = px; pos[n * 3 + 1] = py; pos[n * 3 + 2] = pz;
        int i = n - NPv;
        float dx = px - lig_xyz[i * 3 + 0];
        float dy = py - lig_xyz[i * 3 + 1];
        float dz = pz - lig_xyz[i * 3 + 2];
        atomicAdd(&racc[lig_batch[i]], dx * dx + dy * dy + dz * dz);
    }
    dsum[n * 3 + 0] = 0.f; dsum[n * 3 + 1] = 0.f; dsum[n * 3 + 2] = 0.f;
}

__global__ void rmsd_fin_kernel(float* __restrict__ racc, float* __restrict__ rall, int slot) {
    int b = threadIdx.x;
    if (b >= BBv) return;
    rall[slot * BBv + b] = sqrtf(racc[b] / (float)(NLv / BBv));
    racc[b] = 0.f;
}

__global__ void loss_kernel(const float* __restrict__ rall, float* __restrict__ out) {
    if (threadIdx.x != 0 || blockIdx.x != 0) return;
    float s = 0.f;
    for (int b = 0; b < BBv; b++) {
        float m = 0.f;
        for (int i = 0; i < NLAY; i++) m += rall[i * BBv + b];
        s += m / (float)NLAY + rall[(NREC * NLAY - 1) * BBv + b];
    }
    out[0] = s / (float)BBv;
}

// ---------------------------------------------------------------- WMMA edge MLP
// One wave = 16 edges x 128 outputs. GEMM1: feat[16,385]@Wm1 (12 bf16 WMMA k-steps
// + d2 column via VALU), GEMM2: m[16,128]@Wm2 (4 k-steps), all fp32 accumulate.
__global__ __launch_bounds__(WPB * 32)
void edge_mlp_kernel(const unsigned short* __restrict__ hb,
                     float* __restrict__ e, unsigned short* __restrict__ eb,
                     const float* __restrict__ pos,
                     const int* __restrict__ u, const int* __restrict__ v,
                     const unsigned short* __restrict__ Wm1T,   // [128][K1PAD] bf16
                     const float* __restrict__ Wm1last,         // row 384, f32 [128]
                     const float* __restrict__ bm1,
                     const unsigned short* __restrict__ Wm2T,   // [128][128] bf16
                     const float* __restrict__ bm2,
                     const float* __restrict__ wa,
                     const float* __restrict__ wpos,
                     const float* __restrict__ bpos,
                     float* __restrict__ agg, float* __restrict__ dsum) {
    __shared__ int   u_s[WPB][16], v_s[WPB][16];
    __shared__ float rel_s[WPB][16][3], d2_s[WPB][16];
    __shared__ float a_s[WPB][16], c_s[WPB][16];
    __shared__ __align__(16) unsigned short mbf[WPB][16][DD];   // m1 (bf16) for GEMM2 A
    __shared__ __align__(16) float          mf [WPB][16][DD];   // m2 (f32) for writeback

    const int wid = threadIdx.x >> 5, lane = threadIdx.x & 31;
    const int tile = blockIdx.x * WPB + wid;
    const int e0 = tile * 16;
    const int row = lane & 15;
    const int lo  = (lane < 16) ? 0 : 8;
    const int mb  = (lane < 16) ? 0 : 8;
    const int colq = lane & 15;

    __builtin_prefetch(Wm1T, 0, 3);   // global_prefetch_b8 on the weight panel

    if (lane < 16) {
        int ei = e0 + row;
        int uu = u[ei], vv = v[ei];
        u_s[wid][row] = uu; v_s[wid][row] = vv;
        float rx = pos[uu * 3 + 0] - pos[vv * 3 + 0];
        float ry = pos[uu * 3 + 1] - pos[vv * 3 + 1];
        float rz = pos[uu * 3 + 2] - pos[vv * 3 + 2];
        rel_s[wid][row][0] = rx; rel_s[wid][row][1] = ry; rel_s[wid][row][2] = rz;
        d2_s[wid][row] = rx * rx + ry * ry + rz * rz;
    }
    __syncthreads();

    v8f acc[8];
    for (int i = 0; i < 8; i++) for (int j = 0; j < 8; j++) acc[i][j] = 0.f;

    const int uu = u_s[wid][row], vv = v_s[wid][row];

    // GEMM1: 12 k-steps over [h[u] | h[v] | e]
    for (int kk = 0; kk < 12; kk++) {
        const unsigned short* src;
        if (kk < 4)      src = hb + (size_t)uu * DD;
        else if (kk < 8) src = hb + (size_t)vv * DD;
        else             src = eb + (size_t)(e0 + row) * DD;
        int cb = (kk & 3) * 32;
        Frag a = ldfrag(src + cb + lo, src + cb + lo + 16);
#pragma unroll
        for (int nt = 0; nt < 8; nt++) {
            const unsigned short* wp = Wm1T + (size_t)(nt * 16 + colq) * K1PAD + kk * 32 + lo;
            Frag b = ldfrag(wp, wp + 16);
            acc[nt] = wmma_bf16(a, b, acc[nt]);
        }
    }

    // + d2 * Wm1[384] + bias, SiLU, stage m1 (bf16)
    for (int nt = 0; nt < 8; nt++) {
        int col = nt * 16 + colq;
        float w384 = Wm1last[col], bi = bm1[col];
#pragma unroll
        for (int j = 0; j < 8; j++) {
            float x = acc[nt][j] + d2_s[wid][mb + j] * w384 + bi;
            mbf[wid][mb + j][col] = f2bf(siluf_(x));
        }
    }
    __syncthreads();

    // GEMM2: m1 @ Wm2
    for (int i = 0; i < 8; i++) for (int j = 0; j < 8; j++) acc[i][j] = 0.f;
    for (int kk = 0; kk < 4; kk++) {
        const unsigned short* src = &mbf[wid][row][0];
        int cb = kk * 32;
        Frag a = ldfrag(src + cb + lo, src + cb + lo + 16);
#pragma unroll
        for (int nt = 0; nt < 8; nt++) {
            const unsigned short* wp = Wm2T + (size_t)(nt * 16 + colq) * DD + cb + lo;
            Frag b = ldfrag(wp, wp + 16);
            acc[nt] = wmma_bf16(a, b, acc[nt]);
        }
    }

    // bias + SiLU -> m2; dot with wa / wpos (half-wave shuffle reduction)
    float pa[8], pc[8];
    for (int j = 0; j < 8; j++) { pa[j] = 0.f; pc[j] = 0.f; }
    for (int nt = 0; nt < 8; nt++) {
        int col = nt * 16 + colq;
        float bi = bm2[col], wav = wa[col], wpv = wpos[col];
#pragma unroll
        for (int j = 0; j < 8; j++) {
            float x = siluf_(acc[nt][j] + bi);
            mf[wid][mb + j][col] = x;
            pa[j] += x * wav;
            pc[j] += x * wpv;
        }
    }
    for (int m = 1; m < 16; m <<= 1) {
#pragma unroll
        for (int j = 0; j < 8; j++) {
            pa[j] += __shfl_xor(pa[j], m, 32);
            pc[j] += __shfl_xor(pc[j], m, 32);
        }
    }
    if (colq == 0) {
        float bp = bpos[0];
        for (int j = 0; j < 8; j++) {
            a_s[wid][mb + j] = sigmoidf_(pa[j]);
            c_s[wid][mb + j] = pc[j] + bp;
        }
    }
    __syncthreads();

    // writeback: e += m2; agg[v] += a*m2 (atomics); dsum[v] += rel*coef
    for (int rr = 0; rr < 16; rr++) {
        int ei = e0 + rr;
        int vd = v_s[wid][rr];
        float ar = a_s[wid][rr];
        int c0 = lane * 4;
        float m0 = mf[wid][rr][c0 + 0], m1 = mf[wid][rr][c0 + 1];
        float m2 = mf[wid][rr][c0 + 2], m3 = mf[wid][rr][c0 + 3];
        size_t eo = (size_t)ei * DD + c0;
        float f0 = e[eo + 0] + m0, f1 = e[eo + 1] + m1, f2 = e[eo + 2] + m2, f3 = e[eo + 3] + m3;
        e[eo + 0] = f0; e[eo + 1] = f1; e[eo + 2] = f2; e[eo + 3] = f3;
        eb[eo + 0] = f2bf(f0); eb[eo + 1] = f2bf(f1); eb[eo + 2] = f2bf(f2); eb[eo + 3] = f2bf(f3);
        size_t ao = (size_t)vd * DD + c0;
        atomicAdd(&agg[ao + 0], ar * m0);
        atomicAdd(&agg[ao + 1], ar * m1);
        atomicAdd(&agg[ao + 2], ar * m2);
        atomicAdd(&agg[ao + 3], ar * m3);
    }
    if (lane < 16) {
        int rr = lane;
        int vd = v_s[wid][rr];
        float cf = c_s[wid][rr];
        atomicAdd(&dsum[vd * 3 + 0], rel_s[wid][rr][0] * cf);
        atomicAdd(&dsum[vd * 3 + 1], rel_s[wid][rr][1] * cf);
        atomicAdd(&dsum[vd * 3 + 2], rel_s[wid][rr][2] * cf);
    }
}

// ---------------------------------------------------------------- generic [·,256]@[256,128] WMMA
// mode 0: X = X + A@W + b        (h update, A = [h | agg])
// mode 1: g=sig(A@W+b); X = g*X0 + (1-g)*X   (node gate, A = [h0 | h])
// mode 2: same gating             (edge gate, A = [ei | e])
__global__ __launch_bounds__(WPB * 32)
void pair_gemm_kernel(const unsigned short* __restrict__ A0,
                      const unsigned short* __restrict__ A1,
                      const unsigned short* __restrict__ WT,   // [128][256] bf16
                      const float* __restrict__ bias,
                      const float* __restrict__ X0,
                      float* __restrict__ X, unsigned short* __restrict__ Xb,
                      int mode) {
    const int wid = threadIdx.x >> 5, lane = threadIdx.x & 31;
    const int tile = blockIdx.x * WPB + wid;
    const int r0 = tile * 16;
    const int row = lane & 15;
    const int lo  = (lane < 16) ? 0 : 8;
    const int mb  = (lane < 16) ? 0 : 8;
    const int colq = lane & 15;

    v8f acc[8];
    for (int i = 0; i < 8; i++) for (int j = 0; j < 8; j++) acc[i][j] = 0.f;

    for (int kk = 0; kk < 8; kk++) {
        const unsigned short* src = ((kk < 4) ? A0 : A1) + (size_t)(r0 + row) * DD + (kk & 3) * 32;
        Frag a = ldfrag(src + lo, src + lo + 16);
#pragma unroll
        for (int nt = 0; nt < 8; nt++) {
            const unsigned short* wp = WT + (size_t)(nt * 16 + colq) * 256 + kk * 32 + lo;
            Frag b = ldfrag(wp, wp + 16);
            acc[nt] = wmma_bf16(a, b, acc[nt]);
        }
    }

    for (int nt = 0; nt < 8; nt++) {
        int col = nt * 16 + colq;
        float bi = bias[col];
#pragma unroll
        for (int j = 0; j < 8; j++) {
            size_t off = (size_t)(r0 + mb + j) * DD + col;
            float x;
            if (mode == 0) {
                x = X[off] + acc[nt][j] + bi;
            } else {
                float g = sigmoidf_(acc[nt][j] + bi);
                x = g * X0[off] + (1.f - g) * X[off];
            }
            X[off] = x;
            Xb[off] = f2bf(x);
        }
    }
}

// ---------------------------------------------------------------- host orchestration

extern "C" void kernel_launch(void* const* d_in, const int* in_sizes, int n_in,
                              void* d_out, int out_size, void* d_ws, size_t ws_size,
                              hipStream_t stream) {
    (void)in_sizes; (void)n_in; (void)out_size; (void)ws_size;

    const float* pro_node = (const float*)d_in[0];
    const float* lig_node = (const float*)d_in[1];
    const float* pro_xyz  = (const float*)d_in[2];
    const float* lig_pos  = (const float*)d_in[3];
    const float* lig_xyz  = (const float*)d_in[4];
    const float* p2ps     = (const float*)d_in[5];
    const float* l2ls     = (const float*)d_in[6];
    const float* W_ei     = (const float*)d_in[7];
    const float* b_ei     = (const float*)d_in[8];
    const float* gn_gamma = (const float*)d_in[9];
    const float* gn_beta  = (const float*)d_in[10];
    const float* Wm1      = (const float*)d_in[11];
    const float* bm1      = (const float*)d_in[12];
    const float* Wm2      = (const float*)d_in[13];
    const float* bm2      = (const float*)d_in[14];
    const float* wa       = (const float*)d_in[15];
    const float* Wh       = (const float*)d_in[16];
    const float* bh       = (const float*)d_in[17];
    const float* wpos     = (const float*)d_in[18];
    const float* bpos     = (const float*)d_in[19];
    const float* Wg_node  = (const float*)d_in[20];
    const float* bg_node  = (const float*)d_in[21];
    const float* Wg_edge  = (const float*)d_in[22];
    const float* bg_edge  = (const float*)d_in[23];
    const int* p2p_idx    = (const int*)d_in[24];
    const int* l2l_idx    = (const int*)d_in[25];
    const int* p2l_idx    = (const int*)d_in[26];
    const int* lig_batch  = (const int*)d_in[28];

    char* w = (char*)d_ws;
    auto carve = [&](size_t bytes) -> char* {
        char* p = w;
        w += (bytes + 255) & ~(size_t)255;
        return p;
    };

    float*          h    = (float*)carve((size_t)NN * DD * 4);
    float*          h0   = (float*)carve((size_t)NN * DD * 4);
    unsigned short* hb   = (unsigned short*)carve((size_t)NN * DD * 2);
    unsigned short* h0b  = (unsigned short*)carve((size_t)NN * DD * 2);
    float*          e    = (float*)carve((size_t)EE * DD * 4);
    unsigned short* eb   = (unsigned short*)carve((size_t)EE * DD * 2);
    float*          ei_  = (float*)carve((size_t)EE * DD * 4);
    unsigned short* eib  = (unsigned short*)carve((size_t)EE * DD * 2);
    unsigned short* aggb = (unsigned short*)carve((size_t)NN * DD * 2);
    float*          pos  = (float*)carve((size_t)NN * 3 * 4);
    int*            ue   = (int*)carve((size_t)EE * 4);
    int*            ve   = (int*)carve((size_t)EE * 4);
    unsigned short* Wm1T = (unsigned short*)carve((size_t)NLAY * DD * K1PAD * 2);
    unsigned short* Wm2T = (unsigned short*)carve((size_t)NLAY * DD * DD * 2);
    unsigned short* WhT  = (unsigned short*)carve((size_t)NLAY * DD * 256 * 2);
    unsigned short* WgnT = (unsigned short*)carve((size_t)DD * 256 * 2);
    unsigned short* WgeT = (unsigned short*)carve((size_t)DD * 256 * 2);
    float*          rall = (float*)carve((size_t)NREC * NLAY * BBv * 4);
    float* zreg = (float*)carve(((size_t)NN * DD + NN * 3 + NN + BBv) * 4);
    float* agg    = zreg;
    float* dsum   = zreg + (size_t)NN * DD;
    float* countf = dsum + (size_t)NN * 3;
    float* racc   = countf + NN;

    long zn = (long)NN * DD + NN * 3 + NN + BBv;
    zero_kernel<<<(int)((zn + 255) / 256), 256, 0, stream>>>(zreg, zn);

    {
        long t1 = (long)NLAY * DD * K1PAD;
        transpose_bf16_kernel<<<(int)((t1 + 255) / 256), 256, 0, stream>>>(Wm1, Wm1T, NLAY, 385, DD, K1PAD);
        long t2 = (long)NLAY * DD * DD;
        transpose_bf16_kernel<<<(int)((t2 + 255) / 256), 256, 0, stream>>>(Wm2, Wm2T, NLAY, DD, DD, DD);
        long t3 = (long)NLAY * DD * 256;
        transpose_bf16_kernel<<<(int)((t3 + 255) / 256), 256, 0, stream>>>(Wh, WhT, NLAY, 256, DD, 256);
        long t4 = (long)DD * 256;
        transpose_bf16_kernel<<<(int)((t4 + 255) / 256), 256, 0, stream>>>(Wg_node, WgnT, 1, 256, DD, 256);
        transpose_bf16_kernel<<<(int)((t4 + 255) / 256), 256, 0, stream>>>(Wg_edge, WgeT, 1, 256, DD, 256);
    }

    graphnorm_kernel<<<BBv, DD, 0, stream>>>(pro_node, lig_node, gn_gamma, gn_beta, h, h0, hb, h0b);
    pos_init_kernel<<<(NN + 255) / 256, 256, 0, stream>>>(pro_xyz, lig_pos, pos);
    build_edges_kernel<<<(EE + 255) / 256, 256, 0, stream>>>(p2p_idx, l2l_idx, p2l_idx, ue, ve, countf);
    {
        long t = (long)EE * DD;
        edge_init_kernel<<<(int)((t + 255) / 256), 256, 0, stream>>>(p2ps, l2ls, W_ei, b_ei, ei_, eib, e, eb);
    }

    const int edgeBlocks = EE / 16 / WPB;   // 1040
    const int nodeBlocks = NN / 16 / WPB;   // 40

    for (int r = 0; r < NREC; r++) {
        for (int l = 0; l < NLAY; l++) {
            edge_mlp_kernel<<<edgeBlocks, WPB * 32, 0, stream>>>(
                hb, e, eb, pos, ue, ve,
                Wm1T + (size_t)l * DD * K1PAD,
                Wm1 + ((size_t)l * 385 + 384) * DD,
                bm1 + (size_t)l * DD,
                Wm2T + (size_t)l * DD * DD,
                bm2 + (size_t)l * DD,
                wa + (size_t)l * DD, wpos + (size_t)l * DD, bpos + l,
                agg, dsum);
            cvt_agg_kernel<<<(NN * DD + 255) / 256, 256, 0, stream>>>(agg, aggb);
            pair_gemm_kernel<<<nodeBlocks, WPB * 32, 0, stream>>>(
                hb, aggb, WhT + (size_t)l * DD * 256, bh + (size_t)l * DD, h, h, hb, 0);
            pos_rmsd_kernel<<<(NN + 255) / 256, 256, 0, stream>>>(pos, dsum, countf, lig_xyz, lig_batch, racc);
            rmsd_fin_kernel<<<1, 32, 0, stream>>>(racc, rall, r * NLAY + l);
        }
        pair_gemm_kernel<<<nodeBlocks, WPB * 32, 0, stream>>>(h0b, hb, WgnT, bg_node, h0, h, hb, 1);
        pair_gemm_kernel<<<edgeBlocks, WPB * 32, 0, stream>>>(eib, eb, WgeT, bg_edge, ei_, e, eb, 2);
    }

    loss_kernel<<<1, 32, 0, stream>>>(rall, (float*)d_out);
}
